// GAT_12558484373612
// MI455X (gfx1250) — compile-verified, matrix operand order checked by reading, not run on hardware
//
#include <hip/hip_runtime.h>
#include <hip/hip_bf16.h>

// ---------------------------------------------------------------------------
// GAT 2-layer forward for MI455X (gfx1250, wave32).
//  * Dense projections: v_wmma_f32_16x16x32_bf16, W panel staged into LDS by
//    the Tensor Data Mover (tensor_load_to_lds + s_wait_tensorcnt).
//  * Edge softmax/scatter: memory-bound, wave32-wide (lane == channel, C==32).
// ---------------------------------------------------------------------------

typedef __bf16    v16bf __attribute__((ext_vector_type(16)));
typedef float     v8f   __attribute__((ext_vector_type(8)));
typedef unsigned  u32x4 __attribute__((ext_vector_type(4)));
typedef int       i32x8 __attribute__((ext_vector_type(8)));
typedef int       i32x4 __attribute__((ext_vector_type(4)));

#define NEG_SLOPE 0.2f
#define GAT_EPS   1e-16f

// ----------------------------- WMMA GEMM -----------------------------------
// Y[M,NCOL] = X[M,K] @ W[K,NCOL], row-major. 256 threads = 8 waves per block;
// each wave owns one 16-row tile and sweeps all NCOL/16 column tiles.
// W (K x NCOL f32) is DMA'd into LDS once per workgroup via the TDM.
template <int K, int NCOL>
__global__ void __launch_bounds__(256)
gat_gemm_wmma(const float* __restrict__ X, const float* __restrict__ W,
              float* __restrict__ Y, int M) {
  __shared__ float Wlds[K * NCOL];

  // ---- Tensor Data Mover: load whole W panel (K x NCOL f32) into LDS ----
  {
    const unsigned lds_addr = (unsigned)(uintptr_t)&Wlds[0];  // low 32b = LDS offset
    const unsigned long long ga = (unsigned long long)(uintptr_t)W;
    u32x4 g0;
    g0[0] = 1u;                                        // count=1 (valid user D#)
    g0[1] = lds_addr;                                  // lds_addr
    g0[2] = (unsigned)(ga & 0xffffffffu);              // global_addr[31:0]
    g0[3] = (unsigned)((ga >> 32) & 0x01ffffffu)       // global_addr[56:32]
          | (2u << 30);                                // type=2 ("image")
    i32x8 g1;
    g1[0] = (int)(2u << 16);                           // data_size=2 (4 bytes)
    g1[1] = (int)(((unsigned)NCOL & 0xffffu) << 16);   // tensor_dim0[15:0]
    g1[2] = (int)((((unsigned)NCOL >> 16) & 0xffffu)   // tensor_dim0[31:16]
          | (((unsigned)K & 0xffffu) << 16));          // tensor_dim1[15:0]
    g1[3] = (int)((((unsigned)K >> 16) & 0xffffu)      // tensor_dim1[31:16]
          | (((unsigned)NCOL & 0xffffu) << 16));       // tile_dim0 = NCOL
    g1[4] = (int)((unsigned)K & 0xffffu);              // tile_dim1 = K (tile_dim2=0)
    g1[5] = (int)NCOL;                                 // tensor_dim0_stride lo32
    g1[6] = 0;                                         // stride hi16 | dim1_stride lo
    g1[7] = 0;
    const i32x4 z4 = {0, 0, 0, 0};
    const i32x8 z8 = {0, 0, 0, 0, 0, 0, 0, 0};
    // 6-arg form on this toolchain: (g0, g1, g2, g3, g4, cpol)
    __builtin_amdgcn_tensor_load_to_lds(g0, g1, z4, z4, z8, 0);
    __builtin_amdgcn_s_wait_tensorcnt(0);              // s_wait_tensorcnt 0
  }
  __syncthreads();

  const int lane   = threadIdx.x & 31;
  const int waveId = threadIdx.x >> 5;
  const int tileM  = blockIdx.x * 8 + waveId;
  if (tileM * 16 >= M) return;

  // A fragment layout (16-bit A 16x32): lane half owns K {0..7,16..23} vs
  // {8..15,24..31}; row M = lane%16.
  int arow = tileM * 16 + (lane & 15);
  if (arow >= M) arow = M - 1;                         // M is 16-aligned here
  const int khalf = (lane >> 4) << 3;                  // 0 or 8
  const float* __restrict__ xrow = X + (size_t)arow * K;
  __builtin_prefetch(xrow, 0, 1);                      // global_prefetch_b8

  constexpr int KS = K / 32;
  v16bf afrag[KS];
#pragma unroll
  for (int ks = 0; ks < KS; ++ks) {
#pragma unroll
    for (int j = 0; j < 8; ++j) {
      afrag[ks][j]     = (__bf16)xrow[ks * 32 + khalf + j];
      afrag[ks][8 + j] = (__bf16)xrow[ks * 32 + 16 + khalf + j];
    }
  }

  const int mbase = tileM * 16 + ((lane >> 4) << 3);   // C/D: M = v + 8*(lane/16)
#pragma unroll
  for (int tn = 0; tn < NCOL / 16; ++tn) {
    const int col = tn * 16 + (lane & 15);
    v8f acc = {};
#pragma unroll
    for (int ks = 0; ks < KS; ++ks) {
      v16bf b;                                         // B col-striped from LDS
#pragma unroll
      for (int j = 0; j < 8; ++j) {
        b[j]     = (__bf16)Wlds[(ks * 32 + khalf + j) * NCOL + col];
        b[8 + j] = (__bf16)Wlds[(ks * 32 + 16 + khalf + j) * NCOL + col];
      }
      acc = __builtin_amdgcn_wmma_f32_16x16x32_bf16(
          /*neg_a=*/false, afrag[ks], /*neg_b=*/false, b,
          /*c_mod=*/(short)0, acc, /*reuse_a=*/false, /*reuse_b=*/false);
    }
#pragma unroll
    for (int v = 0; v < 8; ++v) {
      const int r = mbase + v;
      if (r < M) Y[(size_t)r * NCOL + col] = acc[v];
    }
  }
}

// --------------------------- attention logits -------------------------------
__global__ void gat_alpha(const float* __restrict__ hfeat,
                          const float* __restrict__ a_src,
                          const float* __restrict__ a_dst,
                          float* __restrict__ as, float* __restrict__ ad,
                          int N, int H, int C) {
  const int i = blockIdx.x * blockDim.x + threadIdx.x;
  if (i >= N * H) return;
  const int n = i / H, h = i % H;
  const float* hp = hfeat + ((size_t)n * H + h) * C;
  float ss = 0.f, sd = 0.f;
  for (int c = 0; c < C; ++c) {
    const float v = hp[c];
    ss = fmaf(v, a_src[h * C + c], ss);
    sd = fmaf(v, a_dst[h * C + c], sd);
  }
  as[i] = ss;
  ad[i] = sd;
}

// ------------------------- segment softmax helpers ---------------------------
__device__ __forceinline__ unsigned f2key(float f) {
  unsigned b = __float_as_uint(f);
  return (b & 0x80000000u) ? ~b : (b | 0x80000000u);   // order-preserving key
}
__device__ __forceinline__ float key2f(unsigned k) {
  unsigned b = (k & 0x80000000u) ? (k & 0x7fffffffu) : ~k;
  return __uint_as_float(b);
}
__device__ __forceinline__ float lrelu(float v) {
  return v > 0.f ? v : NEG_SLOPE * v;
}

__global__ void gat_seg_init(unsigned* __restrict__ mkey, float* __restrict__ denom,
                             float* __restrict__ acc, int NH, long long NHC) {
  const long long i = (long long)blockIdx.x * blockDim.x + threadIdx.x;
  if (i < NH) { mkey[i] = 0u; denom[i] = 0.f; }
  if (i < NHC) acc[i] = 0.f;
}

// pass 1: segment max of leaky_relu(as[s]+ad[d]) over incoming edges of d
__global__ void gat_edge_max(const int* __restrict__ src, const int* __restrict__ dst,
                             int E, int N,
                             const float* __restrict__ as, const float* __restrict__ ad,
                             unsigned* __restrict__ mkey, int H) {
  const int e = blockIdx.x * blockDim.x + threadIdx.x;
  if (e >= E + N) return;
  const int s = (e < E) ? src[e] : (e - E);
  const int d = (e < E) ? dst[e] : (e - E);
  for (int h = 0; h < H; ++h) {
    const float v = lrelu(as[s * H + h] + ad[d * H + h]);
    atomicMax(&mkey[d * H + h], f2key(v));
  }
}

// pass 2: denom[d,h] += exp(e - m[d,h])
__global__ void gat_edge_sum(const int* __restrict__ src, const int* __restrict__ dst,
                             int E, int N,
                             const float* __restrict__ as, const float* __restrict__ ad,
                             const unsigned* __restrict__ mkey,
                             float* __restrict__ denom, int H) {
  const int e = blockIdx.x * blockDim.x + threadIdx.x;
  if (e >= E + N) return;
  const int s = (e < E) ? src[e] : (e - E);
  const int d = (e < E) ? dst[e] : (e - E);
  for (int h = 0; h < H; ++h) {
    const float v = lrelu(as[s * H + h] + ad[d * H + h]);
    atomicAdd(&denom[d * H + h], __expf(v - key2f(mkey[d * H + h])));
  }
}

// pass 3: acc[d,h*C+c] += h[s,h*C+c] * alpha. One wave per edge, lane == c.
__global__ void gat_edge_msg(const int* __restrict__ src, const int* __restrict__ dst,
                             int E, int N,
                             const float* __restrict__ as, const float* __restrict__ ad,
                             const unsigned* __restrict__ mkey,
                             const float* __restrict__ denom,
                             const float* __restrict__ hfeat,
                             float* __restrict__ acc, int H) {
  const int wid  = (blockIdx.x * blockDim.x + threadIdx.x) >> 5;
  const int lane = threadIdx.x & 31;                   // channel, C == 32
  if (wid >= E + N) return;
  const int s = (wid < E) ? src[wid] : (wid - E);
  const int d = (wid < E) ? dst[wid] : (wid - E);
  for (int h = 0; h < H; ++h) {
    const float v    = lrelu(as[s * H + h] + ad[d * H + h]);
    const float coef = __expf(v - key2f(mkey[d * H + h])) /
                       (denom[d * H + h] + GAT_EPS);
    const size_t so = ((size_t)s * H + h) * 32 + lane;
    const size_t dd = ((size_t)d * H + h) * 32 + lane;
    atomicAdd(&acc[dd], hfeat[so] * coef);
  }
}

// out = relu(acc + b) in place (layer 1 epilogue)
__global__ void gat_bias_relu(float* __restrict__ acc, const float* __restrict__ b,
                              int N, int F) {
  const long long i = (long long)blockIdx.x * blockDim.x + threadIdx.x;
  if (i >= (long long)N * F) return;
  const float v = acc[i] + b[(int)(i % F)];
  acc[i] = v > 0.f ? v : 0.f;
}

// seed layer-2 output accumulator with bias b2 (so final = seg_sum + b2)
__global__ void gat_seed_bias(float* __restrict__ out, const float* __restrict__ b,
                              int N, int F) {
  const long long i = (long long)blockIdx.x * blockDim.x + threadIdx.x;
  if (i >= (long long)N * F) return;
  out[i] = b[(int)(i % F)];
}

// ---------------------------------------------------------------------------
extern "C" void kernel_launch(void* const* d_in, const int* in_sizes, int n_in,
                              void* d_out, int out_size, void* d_ws, size_t ws_size,
                              hipStream_t stream) {
  const float* x      = (const float*)d_in[0];   // [N,128]
  const int*   eidx   = (const int*)  d_in[1];   // [2,E]
  const float* W1     = (const float*)d_in[2];   // [128,128]
  const float* a_src1 = (const float*)d_in[3];   // [4,32]
  const float* a_dst1 = (const float*)d_in[4];
  const float* b1     = (const float*)d_in[5];   // [128]
  const float* W2     = (const float*)d_in[6];   // [128,32]
  const float* a_src2 = (const float*)d_in[7];   // [1,32]
  const float* a_dst2 = (const float*)d_in[8];
  const float* b2     = (const float*)d_in[9];   // [32]

  const int Fin = 128, C = 32;
  const int H1  = in_sizes[3] / C;               // 4
  const int N   = in_sizes[0] / Fin;             // 50000
  const int E   = in_sizes[1] / 2;               // 800000
  const int F1  = H1 * C;                        // 128
  const int Etot = E + N;                        // with self loops

  const int* src = eidx;                         // edge_index[0]
  const int* dst = eidx + E;                     // edge_index[1]

  // ------- workspace carve-out (floats) -------
  float* ws = (float*)d_ws;
  size_t off = 0;
  float*    h1   = ws + off; off += (size_t)N * F1;
  float*    as1  = ws + off; off += (size_t)N * H1;
  float*    ad1  = ws + off; off += (size_t)N * H1;
  unsigned* m1   = (unsigned*)(ws + off); off += (size_t)N * H1;
  float*    den1 = ws + off; off += (size_t)N * H1;
  float*    acc1 = ws + off; off += (size_t)N * F1;   // relu(h) for layer 2
  float*    h2   = ws + off; off += (size_t)N * C;
  float*    as2  = ws + off; off += (size_t)N;
  float*    ad2  = ws + off; off += (size_t)N;
  unsigned* m2   = (unsigned*)(ws + off); off += (size_t)N;
  float*    den2 = ws + off; off += (size_t)N;
  (void)ws_size;

  float* out = (float*)d_out;                    // [N,32]

  const int TB = 256;
  auto blks = [](long long n, int tb) { return (unsigned)((n + tb - 1) / tb); };
  const unsigned gemmBlocks = (unsigned)(((N + 15) / 16 + 7) / 8);

  // =========================== Layer 1 ===========================
  gat_gemm_wmma<128, 128><<<gemmBlocks, TB, 0, stream>>>(x, W1, h1, N);
  gat_alpha<<<blks((long long)N * H1, TB), TB, 0, stream>>>(h1, a_src1, a_dst1,
                                                            as1, ad1, N, H1, C);
  gat_seg_init<<<blks((long long)N * F1, TB), TB, 0, stream>>>(m1, den1, acc1,
                                                               N * H1, (long long)N * F1);
  gat_edge_max<<<blks(Etot, TB), TB, 0, stream>>>(src, dst, E, N, as1, ad1, m1, H1);
  gat_edge_sum<<<blks(Etot, TB), TB, 0, stream>>>(src, dst, E, N, as1, ad1, m1, den1, H1);
  gat_edge_msg<<<blks((long long)Etot * 32, TB), TB, 0, stream>>>(src, dst, E, N,
                                                                  as1, ad1, m1, den1,
                                                                  h1, acc1, H1);
  gat_bias_relu<<<blks((long long)N * F1, TB), TB, 0, stream>>>(acc1, b1, N, F1);

  // =========================== Layer 2 ===========================
  gat_gemm_wmma<128, 32><<<gemmBlocks, TB, 0, stream>>>(acc1, W2, h2, N);
  gat_alpha<<<blks(N, TB), TB, 0, stream>>>(h2, a_src2, a_dst2, as2, ad2, N, 1, C);
  gat_seed_bias<<<blks((long long)N * C, TB), TB, 0, stream>>>(out, b2, N, C);
  gat_seg_init<<<blks(N, TB), TB, 0, stream>>>(m2, den2, den2 /*dummy*/, N, 0);
  gat_edge_max<<<blks(Etot, TB), TB, 0, stream>>>(src, dst, E, N, as2, ad2, m2, 1);
  gat_edge_sum<<<blks(Etot, TB), TB, 0, stream>>>(src, dst, E, N, as2, ad2, m2, den2, 1);
  gat_edge_msg<<<blks((long long)Etot * 32, TB), TB, 0, stream>>>(src, dst, E, N,
                                                                  as2, ad2, m2, den2,
                                                                  h2, out, 1);
}
